// DeepSeekMoE_44427141710498
// MI455X (gfx1250) — compile-verified, matrix operand order checked by reading, not run on hardware
//
#include <hip/hip_runtime.h>
#include <hip/hip_bf16.h>
#include <math.h>

typedef __bf16 bf16;
typedef __attribute__((ext_vector_type(16))) __bf16 v16bf;
typedef __attribute__((ext_vector_type(8)))  __bf16 v8bf;
typedef __attribute__((ext_vector_type(8)))  float  v8f;
typedef __attribute__((ext_vector_type(4)))  int    v4i;

#define NTOK 4096
#define DIM  1024
#define HID  4096
#define NEXP 8
#define NSH  2
#define CH   128   // H-chunk per iteration
#define NW   8     // waves per block
#define MT   32    // token tile (two 16-row WMMA sub-tiles)

// Optional gfx1250 async global->LDS path (guarded; falls back to uint4 copy).
#ifdef __has_builtin
#if __has_builtin(__builtin_amdgcn_global_load_async_to_lds_b128) && \
    __has_builtin(__builtin_amdgcn_s_wait_asynccnt)
#define USE_ASYNC_LDS 1
#endif
#endif
#ifndef USE_ASYNC_LDS
#define USE_ASYNC_LDS 0
#endif

#if USE_ASYNC_LDS
typedef __attribute__((address_space(1))) v4i* gv4i_t;   // global v4i*
typedef __attribute__((address_space(3))) v4i* lv4i_t;   // LDS v4i*
#endif

// ---------------------------------------------------------------------------
// A fragment from LDS, per CDNA5 ISA 7.12.2 (wave32, 16-bit A 16x32):
// lane L: M = L&15; lanes 0-15 elems 0..7 -> K 0..7, elems 8..15 -> K 16..23;
// lanes 16-31 same pattern shifted by +8 in K.  -> two 16B ds_loads per lane.
// ---------------------------------------------------------------------------
static __device__ __forceinline__ v16bf load_a_frag(const bf16* base, int rs, int lane) {
    int m  = lane & 15;
    int kb = (lane >> 4) << 3;           // 0 or 8
    const bf16* p = base + m * rs + kb;
    v8bf lo = *(const v8bf*)(p);
    v8bf hi = *(const v8bf*)(p + 16);
    return __builtin_shufflevector(lo, hi, 0,1,2,3,4,5,6,7,8,9,10,11,12,13,14,15);
}

// B fragment from pre-packed weights: frag = contiguous [32 lanes][16 bf16]
// block (1KB) -> two coalesced global_load_b128 per lane.
static __device__ __forceinline__ v16bf load_b_packed(const bf16* Wp, int frag, int lane) {
    return *(const v16bf*)(Wp + ((size_t)frag * 32 + lane) * 16);
}

static __device__ __forceinline__ v8f wmma_bf16(v16bf a, v16bf b, v8f c) {
    return __builtin_amdgcn_wmma_f32_16x16x32_bf16(false, a, false, b,
                                                   (short)0, c, false, false);
}

// Copy 16 x 16B from global to LDS for the x-tile gather.
static __device__ __forceinline__ void gather_row256(const bf16* src, bf16* dst) {
#if USE_ASYNC_LDS
#pragma unroll
    for (int j = 0; j < 16; ++j)
        __builtin_amdgcn_global_load_async_to_lds_b128(
            (gv4i_t)(src + j * 8), (lv4i_t)(dst + j * 8), 0, 0);
    __builtin_amdgcn_s_wait_asynccnt(0);
#else
    const uint4* s4 = (const uint4*)src;
    uint4* d4 = (uint4*)dst;
#pragma unroll
    for (int j = 0; j < 16; ++j) d4[j] = s4[j];
#endif
}

// ---------------------------------------------------------------------------
// Weight pack: W[e][K][N] f32 (ld = N) -> bf16 fragments
// out[((e*(N/16)+nt)*(K/32)+ks)][lane][16] with the WMMA B-layout.
// ---------------------------------------------------------------------------
__global__ void pack_w_kernel(const float* __restrict__ W, bf16* __restrict__ out,
                              int nexp, int Kd, int Nd) {
    size_t total = (size_t)nexp * (Nd / 16) * (Kd / 32) * 32;
    size_t stride = (size_t)gridDim.x * blockDim.x;
    for (size_t idx = (size_t)blockIdx.x * blockDim.x + threadIdx.x; idx < total;
         idx += stride) {
        int    lane = (int)(idx & 31);
        size_t f    = idx >> 5;
        int    ks   = (int)(f % (Kd / 32));
        size_t f2   = f / (Kd / 32);
        int    nt   = (int)(f2 % (Nd / 16));
        int    e    = (int)(f2 / (Nd / 16));
        int    n    = nt * 16 + (lane & 15);
        int    kb   = (lane >> 4) << 3;
        const float* p = W + (size_t)e * Kd * Nd + (size_t)(ks * 32 + kb) * Nd + n;
        v16bf v;
#pragma unroll
        for (int i = 0; i < 8; ++i) {
            v[i]     = (bf16)p[(size_t)i * Nd];
            v[i + 8] = (bf16)p[(size_t)(16 + i) * Nd];
        }
        *(v16bf*)(out + idx * 16) = v;
    }
}

// ---------------------------------------------------------------------------
// Fused 2-layer FFN, 32-token tile, chunked over H; 1-deep software pipeline
// so each WMMA consumes loads issued one iteration earlier.
// ---------------------------------------------------------------------------
static __device__ __forceinline__ void ffn_pass(v8f acc0[8], v8f acc1[8],
                                                const bf16* xt, bf16* hb,
                                                const bf16* W1p, const float* b1,
                                                const bf16* W2p, int lane, int w) {
    for (int c = 0; c < HID / CH; ++c) {
        __syncthreads();
        // ---- layer 1: h cols [c*CH + w*16, +16), K = DIM, both row halves
        v8f h0 = {0.f,0.f,0.f,0.f,0.f,0.f,0.f,0.f};
        v8f h1 = h0;
        const int fbase = (c * (CH / 16) + w) * (DIM / 32);
        v16bf a0 = load_a_frag(xt, DIM, lane);
        v16bf a1 = load_a_frag(xt + 16 * DIM, DIM, lane);
        v16bf b  = load_b_packed(W1p, fbase, lane);
        for (int kk = 0; kk < DIM - 32; kk += 32) {
            v16bf a0n = load_a_frag(xt + kk + 32, DIM, lane);
            v16bf a1n = load_a_frag(xt + 16 * DIM + kk + 32, DIM, lane);
            v16bf bn  = load_b_packed(W1p, fbase + (kk >> 5) + 1, lane);
            h0 = wmma_bf16(a0, b, h0);
            h1 = wmma_bf16(a1, b, h1);
            a0 = a0n; a1 = a1n; b = bn;
        }
        h0 = wmma_bf16(a0, b, h0);
        h1 = wmma_bf16(a1, b, h1);

        int   hi   = lane >> 4;
        float bias = b1[c * CH + w * 16 + (lane & 15)];
        int   lc   = w * 16 + (lane & 15);
#pragma unroll
        for (int j = 0; j < 8; ++j) {
            int r = j + 8 * hi;
            float v0 = h0[j] + bias; v0 = v0 > 0.f ? v0 : 0.f;
            float v1 = h1[j] + bias; v1 = v1 > 0.f ? v1 : 0.f;
            hb[r * CH + lc]        = (bf16)v0;
            hb[(16 + r) * CH + lc] = (bf16)v1;
        }
        __syncthreads();
        // ---- layer 2: wave D-slice n0 = w*128, K = this 128-chunk of H
        const int kf0 = c * (CH / 32);
        v16bf p0 = load_a_frag(hb, CH, lane);
        v16bf p1 = load_a_frag(hb + 16 * CH, CH, lane);
        v16bf bb = load_b_packed(W2p, (w * 8) * (HID / 32) + kf0, lane);
#pragma unroll
        for (int s = 0; s < CH / 32; ++s) {
            v16bf q0 = p0, q1 = p1;
            if (s + 1 < CH / 32) {
                q0 = load_a_frag(hb + (s + 1) * 32, CH, lane);
                q1 = load_a_frag(hb + 16 * CH + (s + 1) * 32, CH, lane);
            }
#pragma unroll
            for (int nt = 0; nt < 8; ++nt) {
                v16bf bn = bb;
                if (nt < 7)
                    bn = load_b_packed(W2p, (w * 8 + nt + 1) * (HID / 32) + kf0 + s, lane);
                else if (s + 1 < CH / 32)
                    bn = load_b_packed(W2p, (w * 8) * (HID / 32) + kf0 + s + 1, lane);
                acc0[nt] = wmma_bf16(p0, bb, acc0[nt]);
                acc1[nt] = wmma_bf16(p1, bb, acc1[nt]);
                bb = bn;
            }
            p0 = q0; p1 = q1;
        }
    }
}

// ---------------------------------------------------------------------------
__global__ void cvt_x_kernel(const float* __restrict__ in, bf16* __restrict__ out, int n) {
    int i = blockIdx.x * blockDim.x + threadIdx.x;
    int stride = gridDim.x * blockDim.x;
    for (; i < n; i += stride) out[i] = (bf16)in[i];
}

__global__ void zero_cnts_kernel(int* cnts) {
    if (threadIdx.x < NEXP) cnts[threadIdx.x] = 0;
}

// ---------------------------------------------------------------------------
// Noisy top-2 router. One wave per token.
// ---------------------------------------------------------------------------
__global__ void router_kernel(const float* __restrict__ x,
                              const float* __restrict__ Wr, const float* __restrict__ br,
                              const float* __restrict__ Wn, const float* __restrict__ bn,
                              const float* __restrict__ noise,
                              int* __restrict__ cnts, int* __restrict__ list,
                              float* __restrict__ glist) {
    int w    = threadIdx.x >> 5;
    int lane = threadIdx.x & 31;
    int t    = blockIdx.x * NW + w;
    if (t >= NTOK) return;

    float aR[NEXP], aN[NEXP];
#pragma unroll
    for (int e = 0; e < NEXP; ++e) { aR[e] = 0.f; aN[e] = 0.f; }
    const float* xr = x + (size_t)t * DIM;
    for (int d = lane; d < DIM; d += 32) {
        float xv = xr[d];
        const float* wr = Wr + d * NEXP;
        const float* wn = Wn + d * NEXP;
#pragma unroll
        for (int e = 0; e < NEXP; ++e) {
            aR[e] += xv * wr[e];
            aN[e] += xv * wn[e];
        }
    }
#pragma unroll
    for (int off = 16; off > 0; off >>= 1) {
#pragma unroll
        for (int e = 0; e < NEXP; ++e) {
            aR[e] += __shfl_down(aR[e], off, 32);
            aN[e] += __shfl_down(aN[e], off, 32);
        }
    }
    if (lane == 0) {
        float ns[NEXP];
#pragma unroll
        for (int e = 0; e < NEXP; ++e) {
            float lg = aR[e] + br[e];
            float nv = aN[e] + bn[e];
            float sp = (nv > 20.f) ? nv : log1pf(expf(nv));     // softplus
            ns[e] = lg + noise[(size_t)t * NEXP + e] * sp;
        }
        int b1i = 0; float b1v = ns[0];
#pragma unroll
        for (int e = 1; e < NEXP; ++e) if (ns[e] > b1v) { b1v = ns[e]; b1i = e; }
        int b2i = (b1i == 0) ? 1 : 0; float b2v = ns[b2i];
#pragma unroll
        for (int e = 0; e < NEXP; ++e)
            if (e != b1i && ns[e] > b2v) { b2v = ns[e]; b2i = e; }
        float ex = expf(b2v - b1v);
        float g1 = 1.f / (1.f + ex);
        float g2 = ex / (1.f + ex);
        int p1 = atomicAdd(&cnts[b1i], 1);
        list[b1i * NTOK + p1]  = t;                 // slot 0
        glist[b1i * NTOK + p1] = g1;
        int p2 = atomicAdd(&cnts[b2i], 1);
        list[b2i * NTOK + p2]  = t | (1 << 16);     // slot 1
        glist[b2i * NTOK + p2] = g2;
    }
}

// ---------------------------------------------------------------------------
// Routed experts (sparse, gathered 32-token tiles).
// contrib layout: [token][slot(2)][DIM] f32
// ---------------------------------------------------------------------------
__global__ __launch_bounds__(256)
void routed_ffn_kernel(const bf16* __restrict__ xbf,
                       const bf16* __restrict__ W1p, const float* __restrict__ rb1,
                       const bf16* __restrict__ W2p, const float* __restrict__ rb2,
                       const int* __restrict__ cnts, const int* __restrict__ list,
                       const float* __restrict__ glist, float* __restrict__ contrib) {
    __shared__ bf16  xt[MT * DIM];
    __shared__ bf16  hb[MT * CH];
    __shared__ int   tokS[MT];
    __shared__ int   slotS[MT];
    __shared__ int   valS[MT];
    __shared__ float gateS[MT];

    int e = blockIdx.y, tile = blockIdx.x;
    int cnt = cnts[e];
    if (tile * MT >= cnt) return;

    int tid = threadIdx.x;
    if (tid < MT) {
        int idx = tile * MT + tid;
        int valid = idx < cnt;
        int entry = valid ? list[e * NTOK + idx] : 0;
        tokS[tid]  = entry & 0xFFFF;
        slotS[tid] = (entry >> 16) & 1;
        valS[tid]  = valid;
        gateS[tid] = valid ? glist[e * NTOK + idx] : 0.f;
    }
    __syncthreads();

    // gather 32 token rows of bf16 x into LDS (each thread: 256B)
    {
        int r = tid >> 3, seg = tid & 7;
        gather_row256(xbf + (size_t)tokS[r] * DIM + seg * 128,
                      xt + r * DIM + seg * 128);
    }

    int lane = tid & 31, w = tid >> 5;
    v8f acc0[8], acc1[8];
    v8f z = {0.f,0.f,0.f,0.f,0.f,0.f,0.f,0.f};
#pragma unroll
    for (int nt = 0; nt < 8; ++nt) { acc0[nt] = z; acc1[nt] = z; }

    ffn_pass(acc0, acc1, xt, hb,
             W1p + (size_t)e * DIM * HID, rb1 + (size_t)e * HID,
             W2p + (size_t)e * HID * DIM, lane, w);

    int hi = lane >> 4;
#pragma unroll
    for (int nt = 0; nt < 8; ++nt) {
        int n = w * 128 + nt * 16 + (lane & 15);
        float b2v = rb2[(size_t)e * DIM + n];
#pragma unroll
        for (int j = 0; j < 8; ++j) {
            int m0 = j + 8 * hi;
            int m1 = 16 + m0;
            if (valS[m0])
                contrib[((size_t)tokS[m0] * 2 + slotS[m0]) * DIM + n] =
                    (acc0[nt][j] + b2v) * gateS[m0];
            if (valS[m1])
                contrib[((size_t)tokS[m1] * 2 + slotS[m1]) * DIM + n] =
                    (acc1[nt][j] + b2v) * gateS[m1];
        }
    }
}

// ---------------------------------------------------------------------------
// Shared experts (dense) + final combine:
// out = x + contrib[slot0] + contrib[slot1] + sum_e sharedFFN_e(x)
// ---------------------------------------------------------------------------
__global__ __launch_bounds__(256)
void shared_combine_kernel(const float* __restrict__ xf, const bf16* __restrict__ xbf,
                           const bf16* __restrict__ sW1p, const float* __restrict__ sb1,
                           const bf16* __restrict__ sW2p, const float* __restrict__ sb2,
                           const float* __restrict__ contrib, float* __restrict__ out) {
    __shared__ bf16 xt[MT * DIM];
    __shared__ bf16 hb[MT * CH];

    int tid = threadIdx.x;
    int t0  = blockIdx.x * MT;
    {
        int r = tid >> 3, seg = tid & 7;
        gather_row256(xbf + (size_t)(t0 + r) * DIM + seg * 128,
                      xt + r * DIM + seg * 128);
    }

    int lane = tid & 31, w = tid >> 5;
    v8f acc0[8], acc1[8];
    v8f z = {0.f,0.f,0.f,0.f,0.f,0.f,0.f,0.f};
#pragma unroll
    for (int nt = 0; nt < 8; ++nt) { acc0[nt] = z; acc1[nt] = z; }

    for (int e = 0; e < NSH; ++e)
        ffn_pass(acc0, acc1, xt, hb,
                 sW1p + (size_t)e * DIM * HID, sb1 + (size_t)e * HID,
                 sW2p + (size_t)e * HID * DIM, lane, w);

    int hi = lane >> 4;
#pragma unroll
    for (int nt = 0; nt < 8; ++nt) {
        int n = w * 128 + nt * 16 + (lane & 15);
        float b2s = sb2[n] + sb2[DIM + n];
#pragma unroll
        for (int j = 0; j < 8; ++j) {
            int t_0 = t0 + j + 8 * hi;
            int t_1 = t_0 + 16;
            size_t o0 = (size_t)t_0 * DIM + n;
            size_t o1 = (size_t)t_1 * DIM + n;
            out[o0] = xf[o0] + contrib[(size_t)t_0 * 2 * DIM + n]
                    + contrib[(size_t)t_0 * 2 * DIM + DIM + n] + acc0[nt][j] + b2s;
            out[o1] = xf[o1] + contrib[(size_t)t_1 * 2 * DIM + n]
                    + contrib[(size_t)t_1 * 2 * DIM + DIM + n] + acc1[nt][j] + b2s;
        }
    }
}

// ---------------------------------------------------------------------------
extern "C" void kernel_launch(void* const* d_in, const int* in_sizes, int n_in,
                              void* d_out, int out_size, void* d_ws, size_t ws_size,
                              hipStream_t stream) {
    (void)in_sizes; (void)n_in; (void)out_size; (void)ws_size;
    const float* x     = (const float*)d_in[0];
    const float* noise = (const float*)d_in[1];
    /* d_in[2] = topk (fixed 2) */
    const float* Wr  = (const float*)d_in[3];
    const float* br  = (const float*)d_in[4];
    const float* Wn  = (const float*)d_in[5];
    const float* bn  = (const float*)d_in[6];
    const float* rW1 = (const float*)d_in[7];
    const float* rb1 = (const float*)d_in[8];
    const float* rW2 = (const float*)d_in[9];
    const float* rb2 = (const float*)d_in[10];
    const float* sW1 = (const float*)d_in[11];
    const float* sb1 = (const float*)d_in[12];
    const float* sW2 = (const float*)d_in[13];
    const float* sb2 = (const float*)d_in[14];
    float* out = (float*)d_out;

    char* ws = (char*)d_ws;
    size_t off = 0;
    bf16*  xbf     = (bf16*)(ws + off);  off += (size_t)NTOK * DIM * sizeof(bf16);    // 8 MB
    int*   cnts    = (int*)(ws + off);   off += 256;
    int*   list    = (int*)(ws + off);   off += (size_t)NEXP * NTOK * sizeof(int);    // 128 KB
    float* glist   = (float*)(ws + off); off += (size_t)NEXP * NTOK * sizeof(float);  // 128 KB
    float* contrib = (float*)(ws + off); off += (size_t)NTOK * 2 * DIM * sizeof(float); // 32 MB
    bf16*  rW1p    = (bf16*)(ws + off);  off += (size_t)NEXP * DIM * HID * sizeof(bf16); // 64 MB
    bf16*  rW2p    = (bf16*)(ws + off);  off += (size_t)NEXP * HID * DIM * sizeof(bf16); // 64 MB
    bf16*  sW1p    = (bf16*)(ws + off);  off += (size_t)NSH * DIM * HID * sizeof(bf16);  // 16 MB
    bf16*  sW2p    = (bf16*)(ws + off);  off += (size_t)NSH * HID * DIM * sizeof(bf16);  // 16 MB

    cvt_x_kernel<<<1024, 256, 0, stream>>>(x, xbf, NTOK * DIM);
    zero_cnts_kernel<<<1, 32, 0, stream>>>(cnts);
    pack_w_kernel<<<4096, 256, 0, stream>>>(rW1, rW1p, NEXP, DIM, HID);
    pack_w_kernel<<<4096, 256, 0, stream>>>(rW2, rW2p, NEXP, HID, DIM);
    pack_w_kernel<<<1024, 256, 0, stream>>>(sW1, sW1p, NSH, DIM, HID);
    pack_w_kernel<<<1024, 256, 0, stream>>>(sW2, sW2p, NSH, HID, DIM);
    router_kernel<<<NTOK / NW, 32 * NW, 0, stream>>>(x, Wr, br, Wn, bn, noise,
                                                     cnts, list, glist);
    routed_ffn_kernel<<<dim3(NTOK / MT, NEXP), 256, 0, stream>>>(
        xbf, rW1p, rb1, rW2p, rb2, cnts, list, glist, contrib);
    shared_combine_kernel<<<NTOK / MT, 256, 0, stream>>>(
        x, xbf, sW1p, sb1, sW2p, sb2, contrib, out);
}
